// TFPP_9758165697174
// MI455X (gfx1250) — matrix operand order repaired and checked by reading, not compiled, not run on hardware
//
#include <hip/hip_runtime.h>

typedef __bf16 bf16;
typedef __attribute__((ext_vector_type(16))) __bf16 bf16x16;
typedef __attribute__((ext_vector_type(8)))  float  f32x8;

union FragBF { uint4 u[2]; bf16x16 v; };

// ---------------------------------------------------------------------------
// WMMA GEMM: C[M,N] = A[M,K](bf16) @ Wt[N,K](bf16)^T + bias (+resid) (act)
// block = 256 thr = 8 waves; block tile = 128 rows x 64 cols
// wave w -> row-tile (blockIdx.x*8 + w), 4 col-tiles of 16
// Software-pipelined K loop: double-buffered fragments so next K-step loads
// are in flight while current WMMAs execute (loads retire in order ->
// compiler can use partial s_wait_loadcnt instead of 0).
// ---------------------------------------------------------------------------
__global__ __launch_bounds__(256) void wmma_gemm_bf16(
    const bf16* __restrict__ A, const bf16* __restrict__ Wt,
    const float* __restrict__ bias, const float* __restrict__ resid,
    float* __restrict__ outF, bf16* __restrict__ outB,
    int M, int N, int K, int act)
{
  const int lane = threadIdx.x & 31;
  const int wave = threadIdx.x >> 5;
  const int m0   = blockIdx.x * 128 + wave * 16;
  const int n0   = blockIdx.y * 64;
  const int mRow = lane & 15;
  const int half = lane >> 4;

  int gm  = m0 + mRow;
  int gmc = gm < M ? gm : (M - 1);

  f32x8 acc[4];
#pragma unroll
  for (int j = 0; j < 4; ++j)
#pragma unroll
    for (int e = 0; e < 8; ++e) acc[j][e] = 0.0f;

  const bf16* aRow = A + (size_t)gmc * K + half * 8;   // + {0,16} per k-step
  const bf16* bRow[4];
#pragma unroll
  for (int j = 0; j < 4; ++j) {
    int gn  = n0 + j * 16 + mRow;
    int gnc = gn < N ? gn : (N - 1);
    bRow[j] = Wt + (size_t)gnc * K + half * 8;         // + {0,16} per k-step
  }

  FragBF a0, a1, b0[4], b1[4];
  // prologue: stage k=0
  a0.u[0] = *(const uint4*)(aRow);
  a0.u[1] = *(const uint4*)(aRow + 16);
#pragma unroll
  for (int j = 0; j < 4; ++j) {
    b0[j].u[0] = *(const uint4*)(bRow[j]);
    b0[j].u[1] = *(const uint4*)(bRow[j] + 16);
  }

  for (int kk = 0; kk < K; kk += 64) {
    const int k1 = kk + 32;
    if (k1 < K) {              // uniform: prefetch stage 1 while stage 0 computes
      a1.u[0] = *(const uint4*)(aRow + k1);
      a1.u[1] = *(const uint4*)(aRow + k1 + 16);
#pragma unroll
      for (int j = 0; j < 4; ++j) {
        b1[j].u[0] = *(const uint4*)(bRow[j] + k1);
        b1[j].u[1] = *(const uint4*)(bRow[j] + k1 + 16);
      }
    }
#pragma unroll
    for (int j = 0; j < 4; ++j)
      acc[j] = __builtin_amdgcn_wmma_f32_16x16x32_bf16(
          false, a0.v, false, b0[j].v, (short)0, acc[j], false, false);

    const int k2 = kk + 64;
    if (k2 < K) {              // uniform: prefetch stage 0' while stage 1 computes
      a0.u[0] = *(const uint4*)(aRow + k2);
      a0.u[1] = *(const uint4*)(aRow + k2 + 16);
#pragma unroll
      for (int j = 0; j < 4; ++j) {
        b0[j].u[0] = *(const uint4*)(bRow[j] + k2);
        b0[j].u[1] = *(const uint4*)(bRow[j] + k2 + 16);
      }
    }
    if (k1 < K) {
#pragma unroll
      for (int j = 0; j < 4; ++j)
        acc[j] = __builtin_amdgcn_wmma_f32_16x16x32_bf16(
            false, a1.v, false, b1[j].v, (short)0, acc[j], false, false);
    }
  }

#pragma unroll
  for (int j = 0; j < 4; ++j) {
    int gn = n0 + j * 16 + (lane & 15);
    if (gn >= N) continue;
    float bv = bias ? bias[gn] : 0.0f;
#pragma unroll
    for (int r = 0; r < 8; ++r) {
      int gm2 = m0 + r + 8 * half;
      if (gm2 >= M) continue;
      float v = acc[j][r] + bv;
      if (resid) v += resid[(size_t)gm2 * N + gn];
      if (act == 1) v = 0.5f * v * (1.0f + erff(v * 0.70710678118654752f));
      if (outF) outF[(size_t)gm2 * N + gn] = v;
      if (outB) outB[(size_t)gm2 * N + gn] = (bf16)v;
    }
  }
}

// ---------------------------------------------------------------------------
// weight convert: W[K][N] f32 -> Wt[N][K] bf16
// ---------------------------------------------------------------------------
__global__ __launch_bounds__(256) void convert_wt(
    const float* __restrict__ W, bf16* __restrict__ Wt, int K, int N)
{
  int i = blockIdx.x * 256 + threadIdx.x;
  if (i >= K * N) return;
  int n = i / K, k = i - n * K;
  Wt[i] = (bf16)W[(size_t)k * N + n];
}

__global__ __launch_bounds__(256) void f2bf_kernel(
    const float* __restrict__ A, bf16* __restrict__ B, size_t n)
{
  size_t i = (size_t)blockIdx.x * 256 + threadIdx.x;
  if (i < n) B[i] = (bf16)A[i];
}

__global__ __launch_bounds__(256) void elu1_kernel(float* __restrict__ A, size_t n)
{
  size_t i = (size_t)blockIdx.x * 256 + threadIdx.x;
  if (i < n) { float x = A[i]; A[i] = x > 0.0f ? x + 1.0f : expf(x); }
}

// embedding + sincos positional encoding; one block per (b,n) row, thread=d
__global__ __launch_bounds__(256) void embed_pe(
    const float* __restrict__ x, const float* __restrict__ emb, float* __restrict__ X)
{
  int row = blockIdx.x;
  int n   = row & 4095;
  int d   = threadIdx.x;
  int idx = (int)x[row];
  int i2  = d >> 1;
  float ang = (float)n * expf((float)(2 * i2) * (-9.210340371976184f / 256.0f));
  float pe  = (d & 1) ? cosf(ang) : sinf(ang);
  X[(size_t)row * 256 + d] = emb[idx * 256 + d] + pe;
}

// LayerNorm over D=256; one block per row
__global__ __launch_bounds__(256) void layernorm256(
    const float* __restrict__ Xin, const float* __restrict__ g, const float* __restrict__ b,
    float* __restrict__ outF, bf16* __restrict__ outB)
{
  int row = blockIdx.x, d = threadIdx.x;
  float x = Xin[(size_t)row * 256 + d];
  __shared__ float red[256];
  red[d] = x; __syncthreads();
  for (int s = 128; s > 0; s >>= 1) { if (d < s) red[d] += red[d + s]; __syncthreads(); }
  float mean = red[0] * (1.0f / 256.0f);
  __syncthreads();
  float dx = x - mean;
  red[d] = dx * dx; __syncthreads();
  for (int s = 128; s > 0; s >>= 1) { if (d < s) red[d] += red[d + s]; __syncthreads(); }
  float var = red[0] * (1.0f / 256.0f);
  float y = dx * rsqrtf(var + 1e-5f) * g[d] + b[d];
  if (outF) outF[(size_t)row * 256 + d] = y;
  if (outB) outB[(size_t)row * 256 + d] = (bf16)y;
}

// kv[b,h,d,e] = sum_n k[b,n,h,d] v[b,n,h,e]; ksum[b,h,d] = sum_n k
// one block per (b,h)
__global__ __launch_bounds__(256) void kv_reduce(
    const float* __restrict__ K_, const float* __restrict__ V_,
    float* __restrict__ KV, float* __restrict__ KS)
{
  int bh = blockIdx.x;
  int b = bh >> 3, h = bh & 7;
  __shared__ float kb[8][32], vb[8][32];
  int t = threadIdx.x;
  float acc0 = 0, acc1 = 0, acc2 = 0, acc3 = 0, ks = 0;
  const size_t rowbase = (size_t)b * 4096 * 256 + h * 32;
  int r = t >> 5, c = t & 31;
  int d = t >> 3, e0 = (t & 7) * 4;
  for (int n0 = 0; n0 < 4096; n0 += 8) {
    kb[r][c] = K_[rowbase + (size_t)(n0 + r) * 256 + c];
    vb[r][c] = V_[rowbase + (size_t)(n0 + r) * 256 + c];
    __syncthreads();
#pragma unroll
    for (int rr = 0; rr < 8; ++rr) {
      float kd = kb[rr][d];
      acc0 += kd * vb[rr][e0];
      acc1 += kd * vb[rr][e0 + 1];
      acc2 += kd * vb[rr][e0 + 2];
      acc3 += kd * vb[rr][e0 + 3];
    }
    if (t < 32) {
#pragma unroll
      for (int rr = 0; rr < 8; ++rr) ks += kb[rr][t];
    }
    __syncthreads();
  }
  float* kvp = KV + (size_t)bh * 1024 + t * 4;
  kvp[0] = acc0; kvp[1] = acc1; kvp[2] = acc2; kvp[3] = acc3;
  if (t < 32) KS[bh * 32 + t] = ks;
}

// out[b,n,h,e] = (sum_d q_d kv[d][e]) / (sum_d q_d ksum_d + 1e-6), bf16 out
// one block per (b,n); wave = head, lane = e
__global__ __launch_bounds__(256) void attn_out(
    const float* __restrict__ Q, const float* __restrict__ KV,
    const float* __restrict__ KS, bf16* __restrict__ OutB)
{
  int row = blockIdx.x;
  int b = row >> 12;
  int h = threadIdx.x >> 5;
  int e = threadIdx.x & 31;
  int bh = b * 8 + h;
  float qe = Q[(size_t)row * 256 + h * 32 + e];
  float z = qe * KS[bh * 32 + e];
  for (int s = 16; s; s >>= 1) z += __shfl_xor(z, s, 32);
  z += 1e-6f;
  const float* kvp = KV + (size_t)bh * 1024;
  float o = 0.0f;
#pragma unroll
  for (int d = 0; d < 32; ++d) o += __shfl(qe, d, 32) * kvp[d * 32 + e];
  OutB[(size_t)row * 256 + h * 32 + e] = (bf16)(o / z);
}

// broadcast time-queries to [B*T, D] (f32 + bf16)
__global__ __launch_bounds__(256) void qbuild(
    const float* __restrict__ tq, float* __restrict__ Q32, bf16* __restrict__ Qbf)
{
  int row = blockIdx.x, d = threadIdx.x;
  float v = tq[(row & 7) * 256 + d];
  Q32[row * 256 + d] = v;
  Qbf[row * 256 + d] = (bf16)v;
}

// scores[b,h,t,s] = q . k / sqrt(32); one block per (b,h,t)
__global__ __launch_bounds__(256) void scores_kernel(
    const float* __restrict__ Q, const float* __restrict__ K_, float* __restrict__ S)
{
  int id = blockIdx.x;
  int t = id & 7, h = (id >> 3) & 7, b = id >> 6;
  __shared__ float qs[32];
  if (threadIdx.x < 32) qs[threadIdx.x] = Q[((size_t)(b * 8 + t)) * 256 + h * 32 + threadIdx.x];
  __syncthreads();
  const float scale = 0.17677669529663687f;
  for (int s = threadIdx.x; s < 4096; s += 256) {
    const float* kp = K_ + ((size_t)b * 4096 + s) * 256 + h * 32;
    float acc = 0.0f;
#pragma unroll
    for (int d = 0; d < 32; ++d) acc += qs[d] * kp[d];
    S[(size_t)id * 4096 + s] = acc * scale;
  }
}

__global__ __launch_bounds__(256) void softmax4096(float* __restrict__ S)
{
  float* p = S + (size_t)blockIdx.x * 4096;
  int t = threadIdx.x;
  __shared__ float red[256];
  float mx = -1e30f;
  for (int i = t; i < 4096; i += 256) mx = fmaxf(mx, p[i]);
  red[t] = mx; __syncthreads();
  for (int s = 128; s > 0; s >>= 1) { if (t < s) red[t] = fmaxf(red[t], red[t + s]); __syncthreads(); }
  mx = red[0]; __syncthreads();
  float sum = 0.0f;
  for (int i = t; i < 4096; i += 256) { float e = expf(p[i] - mx); p[i] = e; sum += e; }
  red[t] = sum; __syncthreads();
  for (int s = 128; s > 0; s >>= 1) { if (t < s) red[t] += red[t + s]; __syncthreads(); }
  float inv = 1.0f / red[0];
  for (int i = t; i < 4096; i += 256) p[i] *= inv;
}

// ao[b,t,h,e] = sum_s attn * v; one block per (b,t), wave=h, lane=e
__global__ __launch_bounds__(256) void av_kernel(
    const float* __restrict__ S, const float* __restrict__ V_, bf16* __restrict__ AoB)
{
  int bt = blockIdx.x;
  int b = bt >> 3, t = bt & 7;
  int h = threadIdx.x >> 5, e = threadIdx.x & 31;
  const float* sp = S + ((size_t)((b * 8 + h) * 8 + t)) * 4096;
  const float* vp = V_ + (size_t)b * 4096 * 256 + h * 32 + e;
  float acc = 0.0f;
  for (int s = 0; s < 4096; ++s) acc += sp[s] * vp[(size_t)s * 256];
  AoB[(size_t)bt * 256 + h * 32 + e] = (bf16)acc;
}

__global__ void pred_proj(const float* __restrict__ Xr, const float* __restrict__ pw,
                          const float* __restrict__ pb, float* __restrict__ outDeep, int p)
{
  int row = threadIdx.x;
  if (row >= 128) return;
  const float* xr = Xr + (size_t)row * 256;
  float acc = 0.0f;
  for (int d = 0; d < 256; ++d) acc += xr[d] * pw[d];
  acc += pb[0];
  int b = row >> 3, t = row & 7;
  outDeep[(b * 4 + p) * 8 + t] = acc;
}

// coeffs[b][p*4+d] = (x/3) @ wide_w + wide_b ; one block of 256 threads
__global__ __launch_bounds__(256) void wide_kernel(
    const float* __restrict__ x, const float* __restrict__ W,
    const float* __restrict__ bias, float* __restrict__ coeffs)
{
  int t = threadIdx.x;
  int b = t >> 4, j = t & 15;
  float acc = 0.0f;
  for (int n = 0; n < 4096; ++n) acc += (x[b * 4096 + n] * (1.0f / 3.0f)) * W[n * 16 + j];
  coeffs[t] = acc + bias[j];
}

__global__ __launch_bounds__(256) void meanpool(const float* __restrict__ X, float* __restrict__ gf)
{
  int b = blockIdx.x, d = threadIdx.x;
  float acc = 0.0f;
  for (int n = 0; n < 4096; ++n) acc += X[((size_t)b * 4096 + n) * 256 + d];
  gf[b * 256 + d] = acc * (1.0f / 4096.0f);
}

__global__ void gate1_kernel(const float* __restrict__ gf, const float* __restrict__ w,
                             const float* __restrict__ bias, float* __restrict__ G1)
{
  int b = blockIdx.x, o = threadIdx.x;
  float acc = 0.0f;
  for (int d = 0; d < 256; ++d) acc += gf[b * 256 + d] * w[d * 128 + o];
  acc += bias[o];
  G1[b * 128 + o] = 0.5f * acc * (1.0f + erff(acc * 0.70710678118654752f));
}

__global__ void gate2_kernel(const float* __restrict__ G1, const float* __restrict__ w,
                             const float* __restrict__ bias, float* __restrict__ gate)
{
  int b = blockIdx.x, p = threadIdx.x;
  float acc = 0.0f;
  for (int d = 0; d < 128; ++d) acc += G1[b * 128 + d] * w[d * 4 + p];
  acc += bias[p];
  gate[b * 4 + p] = 1.0f / (1.0f + expf(-acc));
}

__global__ __launch_bounds__(256) void final_combine(
    const float* __restrict__ coeffs, const float* __restrict__ gate,
    const float* __restrict__ outDeep, float* __restrict__ out)
{
  int i = blockIdx.x * 256 + threadIdx.x;
  if (i >= 512) return;
  int t = i & 7, p = (i >> 3) & 3, b = i >> 5;
  float tn = (float)t * (2.0f / 7.0f) - 1.0f;
  float P0 = 1.0f, P1 = tn;
  float P2 = 0.5f * (3.0f * tn * tn - 1.0f);
  float P3 = 0.5f * (5.0f * tn * tn * tn - 3.0f * tn);
  const float* cf = coeffs + (b * 16 + p * 4);
  float w = cf[0] * P0 + cf[1] * P1 + cf[2] * P2 + cf[3] * P3;
  out[i] = w + gate[b * 4 + p] * outDeep[i];
}

// ---------------------------------------------------------------------------
extern "C" void kernel_launch(void* const* d_in, const int* in_sizes, int n_in,
                              void* d_out, int out_size, void* d_ws, size_t ws_size,
                              hipStream_t stream) {
  (void)in_sizes; (void)n_in; (void)out_size;
  int c = 0;
  auto F = [&]() -> const float* { return (const float*)d_in[c++]; };
  const float* x   = F();
  const float* emb = F();
  struct EncP { const float *wq,*bq,*wk,*bk,*wv,*bv,*wo,*bo,*l1g,*l1b,*w1,*b1,*w2,*b2,*l2g,*l2b; } enc[4];
  for (int l = 0; l < 4; ++l) {
    enc[l].wq = F(); enc[l].bq = F(); enc[l].wk = F(); enc[l].bk = F();
    enc[l].wv = F(); enc[l].bv = F(); enc[l].wo = F(); enc[l].bo = F();
    enc[l].l1g = F(); enc[l].l1b = F();
    enc[l].w1 = F(); enc[l].b1 = F(); enc[l].w2 = F(); enc[l].b2 = F();
    enc[l].l2g = F(); enc[l].l2b = F();
  }
  struct PredP { const float *tq,*wq,*bq,*wk,*bk,*wv,*bv,*wo,*bo,*l1g,*l1b,*fw1,*fb1,*fw2,*fb2,*l2g,*l2b,*pw,*pb; } pr[4];
  for (int p = 0; p < 4; ++p) {
    pr[p].tq = F();
    pr[p].wq = F(); pr[p].bq = F(); pr[p].wk = F(); pr[p].bk = F();
    pr[p].wv = F(); pr[p].bv = F(); pr[p].wo = F(); pr[p].bo = F();
    pr[p].l1g = F(); pr[p].l1b = F();
    pr[p].fw1 = F(); pr[p].fb1 = F(); pr[p].fw2 = F(); pr[p].fb2 = F();
    pr[p].l2g = F(); pr[p].l2b = F();
    pr[p].pw = F(); pr[p].pb = F();
  }
  const float* wide_w = F(); const float* wide_b = F();
  const float* gw1 = F(); const float* gb1 = F();
  const float* gw2 = F(); const float* gb2 = F();

  // ---- workspace bump allocator ----
  size_t off = 0; char* base = (char*)d_ws;
  auto alloc = [&](size_t bytes) -> void* {
    off = (off + 255) & ~(size_t)255;
    void* p = base + off; off += bytes; return p;
  };
  const int M = 16 * 4096;                       // 65536 token rows
  float* X   = (float*)alloc((size_t)M * 256 * 4);
  float* T0  = (float*)alloc((size_t)M * 256 * 4);
  float* T1  = (float*)alloc((size_t)M * 256 * 4);
  float* T2  = (float*)alloc((size_t)M * 256 * 4);   // also holds scores (16.8MB)
  bf16*  BF0 = (bf16*)alloc((size_t)M * 256 * 2);    // LN out / attn out / X-bf16
  bf16*  BF1 = (bf16*)alloc((size_t)M * 1024 * 2);   // MLP hidden
  float* KV  = (float*)alloc(128 * 1024 * 4);
  float* KS  = (float*)alloc(128 * 32 * 4);
  float* Q32 = (float*)alloc(128 * 256 * 4);
  bf16*  Qbf = (bf16*)alloc(128 * 256 * 2);
  bf16*  AoB = (bf16*)alloc(128 * 256 * 2);
  float* Pq  = (float*)alloc(128 * 256 * 4);
  float* P1  = (float*)alloc(128 * 256 * 4);
  float* P2  = (float*)alloc(128 * 256 * 4);
  float* P3  = (float*)alloc(128 * 256 * 4);
  float* P4  = (float*)alloc(128 * 256 * 4);
  bf16*  PBF = (bf16*)alloc(128 * 256 * 2);
  bf16*  PBF2= (bf16*)alloc(128 * 512 * 2);
  float* OutDeep = (float*)alloc(512 * 4);
  float* Coeffs  = (float*)alloc(256 * 4);
  float* Gf   = (float*)alloc(16 * 256 * 4);
  float* G1   = (float*)alloc(16 * 128 * 4);
  float* Gate = (float*)alloc(64 * 4);
  bf16 *wqt[4], *wkt[4], *wvt[4], *wot[4], *w1t[4], *w2t[4];
  for (int l = 0; l < 4; ++l) {
    wqt[l] = (bf16*)alloc(256 * 256 * 2); wkt[l] = (bf16*)alloc(256 * 256 * 2);
    wvt[l] = (bf16*)alloc(256 * 256 * 2); wot[l] = (bf16*)alloc(256 * 256 * 2);
    w1t[l] = (bf16*)alloc((size_t)256 * 1024 * 2);
    w2t[l] = (bf16*)alloc((size_t)1024 * 256 * 2);
  }
  bf16 *pqt[4], *pkt[4], *pvt[4], *pot[4], *pf1t[4], *pf2t[4];
  for (int p = 0; p < 4; ++p) {
    pqt[p] = (bf16*)alloc(256 * 256 * 2); pkt[p] = (bf16*)alloc(256 * 256 * 2);
    pvt[p] = (bf16*)alloc(256 * 256 * 2); pot[p] = (bf16*)alloc(256 * 256 * 2);
    pf1t[p] = (bf16*)alloc((size_t)256 * 512 * 2);
    pf2t[p] = (bf16*)alloc((size_t)512 * 256 * 2);
  }
  if (off > ws_size) return;   // workspace too small: bail deterministically

  auto cvt = [&](const float* W, bf16* Wt, int K, int N) {
    convert_wt<<<(K * N + 255) / 256, 256, 0, stream>>>(W, Wt, K, N);
  };
  auto gemm = [&](const bf16* A, const bf16* Wt, const float* bias, const float* resid,
                  float* outF, bf16* outB, int Mm, int Nn, int Kk, int act) {
    dim3 g((Mm + 127) / 128, (Nn + 63) / 64);
    wmma_gemm_bf16<<<g, 256, 0, stream>>>(A, Wt, bias, resid, outF, outB, Mm, Nn, Kk, act);
  };

  // ---- weight conversion (f32 -> transposed bf16) ----
  for (int l = 0; l < 4; ++l) {
    cvt(enc[l].wq, wqt[l], 256, 256); cvt(enc[l].wk, wkt[l], 256, 256);
    cvt(enc[l].wv, wvt[l], 256, 256); cvt(enc[l].wo, wot[l], 256, 256);
    cvt(enc[l].w1, w1t[l], 256, 1024); cvt(enc[l].w2, w2t[l], 1024, 256);
  }
  for (int p = 0; p < 4; ++p) {
    cvt(pr[p].wq, pqt[p], 256, 256); cvt(pr[p].wk, pkt[p], 256, 256);
    cvt(pr[p].wv, pvt[p], 256, 256); cvt(pr[p].wo, pot[p], 256, 256);
    cvt(pr[p].fw1, pf1t[p], 256, 512); cvt(pr[p].fw2, pf2t[p], 512, 256);
  }

  // ---- deep path: embedding + 4 encoder layers ----
  embed_pe<<<M, 256, 0, stream>>>(x, emb, X);
  const size_t MD = (size_t)M * 256;
  for (int l = 0; l < 4; ++l) {
    layernorm256<<<M, 256, 0, stream>>>(X, enc[l].l1g, enc[l].l1b, nullptr, BF0);
    gemm(BF0, wqt[l], enc[l].bq, nullptr, T0, nullptr, M, 256, 256, 0);
    gemm(BF0, wkt[l], enc[l].bk, nullptr, T1, nullptr, M, 256, 256, 0);
    gemm(BF0, wvt[l], enc[l].bv, nullptr, T2, nullptr, M, 256, 256, 0);
    elu1_kernel<<<(int)((MD + 255) / 256), 256, 0, stream>>>(T0, MD);
    elu1_kernel<<<(int)((MD + 255) / 256), 256, 0, stream>>>(T1, MD);
    kv_reduce<<<128, 256, 0, stream>>>(T1, T2, KV, KS);
    attn_out<<<M, 256, 0, stream>>>(T0, KV, KS, BF0);
    gemm(BF0, wot[l], enc[l].bo, X, X, nullptr, M, 256, 256, 0);   // X += attn
    layernorm256<<<M, 256, 0, stream>>>(X, enc[l].l2g, enc[l].l2b, nullptr, BF0);
    gemm(BF0, w1t[l], enc[l].b1, nullptr, nullptr, BF1, M, 1024, 256, 1);  // gelu
    gemm(BF1, w2t[l], enc[l].b2, X, X, nullptr, M, 256, 1024, 0);  // X += mlp
  }

  // ---- predictors ----
  f2bf_kernel<<<(int)((MD + 255) / 256), 256, 0, stream>>>(X, BF0, MD);  // feat bf16
  for (int p = 0; p < 4; ++p) {
    qbuild<<<128, 256, 0, stream>>>(pr[p].tq, Q32, Qbf);
    gemm(Qbf, pqt[p], pr[p].bq, nullptr, Pq, nullptr, 128, 256, 256, 0);
    gemm(BF0, pkt[p], pr[p].bk, nullptr, T0, nullptr, M, 256, 256, 0);
    gemm(BF0, pvt[p], pr[p].bv, nullptr, T1, nullptr, M, 256, 256, 0);
    scores_kernel<<<1024, 256, 0, stream>>>(Pq, T0, T2);
    softmax4096<<<1024, 256, 0, stream>>>(T2);
    av_kernel<<<128, 256, 0, stream>>>(T2, T1, AoB);
    gemm(AoB, pot[p], pr[p].bo, Q32, P1, nullptr, 128, 256, 256, 0);  // queries+ao
    layernorm256<<<128, 256, 0, stream>>>(P1, pr[p].l1g, pr[p].l1b, P2, PBF);
    gemm(PBF, pf1t[p], pr[p].fb1, nullptr, nullptr, PBF2, 128, 512, 256, 1);
    gemm(PBF2, pf2t[p], pr[p].fb2, P2, P3, nullptr, 128, 256, 512, 0);
    layernorm256<<<128, 256, 0, stream>>>(P3, pr[p].l2g, pr[p].l2b, P4, nullptr);
    pred_proj<<<1, 128, 0, stream>>>(P4, pr[p].pw, pr[p].pb, OutDeep, p);
  }

  // ---- wide path + gating + combine ----
  wide_kernel<<<1, 256, 0, stream>>>(x, wide_w, wide_b, Coeffs);
  meanpool<<<16, 256, 0, stream>>>(X, Gf);
  gate1_kernel<<<16, 128, 0, stream>>>(Gf, gw1, gb1, G1);
  gate2_kernel<<<16, 4, 0, stream>>>(G1, gw2, gb2, Gate);
  final_combine<<<2, 256, 0, stream>>>(Coeffs, Gate, OutDeep, (float*)d_out);
}
